// QNet_76398878261249
// MI455X (gfx1250) — matrix operand order repaired and checked by reading, not compiled, not run on hardware
//
#include <hip/hip_runtime.h>
#include <hip/hip_bf16.h>
#include <math.h>

typedef __attribute__((ext_vector_type(16))) _Float16 v16h;
typedef __attribute__((ext_vector_type(8)))  float    v8f;

#define N_QUBITS 8
#define N_LAYERS 4
#define TT 64
#define HID 8
#define PI_F 3.14159265358979f

// ============================================================================
// Kernel 1: U[M,8] = X[M,K] @ W[8,K]^T + bias   (f32 I/O, f16 WMMA math)
// One wave computes one 16-row x 16-col tile (cols 8..15 are zero padding).
// A-frag layout (16-bit A 16x32, ISA 7.12.2): lane lo=row, hi selects K-half.
// B-frag layout (32x16): lane lo=col n, hi selects K 0-7,16-23 / 8-15,24-31.
// Full K-chunks use unconditional (non-temporal) loads; only the final
// partial chunk masks with clamped addresses + selects (no branches).
// ============================================================================
__global__ __launch_bounds__(256) void proj_wmma_kernel(
    const float* __restrict__ X, const float* __restrict__ W,
    const float* __restrict__ bias, float* __restrict__ U,
    int M, int K)
{
    const int lane = threadIdx.x & 31;
    const int wave = threadIdx.x >> 5;
    const int hi   = lane >> 4;
    const int lo   = lane & 15;
    const int row_base = (blockIdx.x * 8 + wave) * 16;

    // A row for this lane (clamped; out-of-range rows never stored)
    const int rowA = min(row_base + lo, M - 1);
    const float* __restrict__ Xrow = X + (size_t)rowA * K;
    // B column: lanes 8..15 alias a valid row and are masked to zero
    const int n = lo & 7;
    const float* __restrict__ Wrow = W + (size_t)n * K;

    v8f acc = {};

    const int kfull = K >> 5;            // number of complete 32-wide chunks
    for (int kc = 0; kc < kfull; ++kc) {
        const int kbase = kc * 32;
        v16h a, bfrag;
        #pragma unroll
        for (int v = 0; v < 8; ++v) {
            const int k0 = kbase + 2 * v + (v >= 4 ? 8 : 0) + hi * 8;
            a[2 * v    ] = (_Float16)__builtin_nontemporal_load(Xrow + k0);
            a[2 * v + 1] = (_Float16)__builtin_nontemporal_load(Xrow + k0 + 1);
        }
        #pragma unroll
        for (int h = 0; h < 16; ++h) {
            const int k = kbase + hi * 8 + h + (h >= 8 ? 8 : 0);
            const float wv = Wrow[k];
            bfrag[h] = (_Float16)((lo < HID) ? wv : 0.0f);
        }
        acc = __builtin_amdgcn_wmma_f32_16x16x32_f16(
                  false, a, false, bfrag, (short)0, acc, false, false);
    }

    if ((kfull << 5) < K) {              // tail chunk, branchless masking
        const int kbase = kfull * 32;
        v16h a, bfrag;
        #pragma unroll
        for (int v = 0; v < 8; ++v) {
            const int k0 = kbase + 2 * v + (v >= 4 ? 8 : 0) + hi * 8;
            const float x0 = __builtin_nontemporal_load(Xrow + min(k0,     K - 1));
            const float x1 = __builtin_nontemporal_load(Xrow + min(k0 + 1, K - 1));
            a[2 * v    ] = (_Float16)((k0     < K) ? x0 : 0.0f);
            a[2 * v + 1] = (_Float16)((k0 + 1 < K) ? x1 : 0.0f);
        }
        #pragma unroll
        for (int h = 0; h < 16; ++h) {
            const int k = kbase + hi * 8 + h + (h >= 8 ? 8 : 0);
            const float wv = Wrow[min(k, K - 1)];
            bfrag[h] = (_Float16)((lo < HID && k < K) ? wv : 0.0f);
        }
        acc = __builtin_amdgcn_wmma_f32_16x16x32_f16(
                  false, a, false, bfrag, (short)0, acc, false, false);
    }

    // ---- C layout: VGPR r -> (row_base + hi*8 + r, col lo) ----
    if (lo < HID) {
        const float bn = bias[lo];
        #pragma unroll
        for (int r = 0; r < 8; ++r) {
            const int row = row_base + hi * 8 + r;
            if (row < M) U[(size_t)row * HID + lo] = acc[r] + bn;
        }
    }
}

// ============================================================================
// Kernel 2: tanh-RNN recurrence over T + output projection + activation.
// One thread = one batch row. Weights staged in LDS.
// act_mode: 0 -> relu, 1 -> sigmoid * pi. Writes x_p[b, col_off .. col_off+7].
// ============================================================================
__global__ __launch_bounds__(256) void rnn_proj_kernel(
    const float* __restrict__ U,    // [B*T, 8] precomputed x@Wih.T + bih
    const float* __restrict__ Whh, const float* __restrict__ bhh,
    const float* __restrict__ Wp,  const float* __restrict__ bp,
    float* __restrict__ xp,        // [B, 24]
    int act_mode, int col_off, int B)
{
    __shared__ float sWhh[64], sWp[64], sbhh[8], sbp[8];
    const int tid = threadIdx.x;
    if (tid < 64) { sWhh[tid] = Whh[tid]; sWp[tid] = Wp[tid]; }
    if (tid < 8)  { sbhh[tid] = bhh[tid]; sbp[tid] = bp[tid]; }
    __syncthreads();

    const int b = blockIdx.x * blockDim.x + tid;
    if (b >= B) return;

    float h[HID];
    #pragma unroll
    for (int i = 0; i < HID; ++i) h[i] = 0.0f;

    const float* ub = U + (size_t)b * TT * HID;
    for (int t = 0; t < TT; ++t) {
        float nh[HID];
        #pragma unroll
        for (int i = 0; i < HID; ++i) {
            float s = ub[t * HID + i] + sbhh[i];
            #pragma unroll
            for (int j = 0; j < HID; ++j) s += sWhh[i * HID + j] * h[j];
            nh[i] = tanhf(s);
        }
        #pragma unroll
        for (int i = 0; i < HID; ++i) h[i] = nh[i];
    }
    #pragma unroll
    for (int i = 0; i < HID; ++i) {
        float s = sbp[i];
        #pragma unroll
        for (int j = 0; j < HID; ++j) s += sWp[i * HID + j] * h[j];
        float o = (act_mode == 0) ? fmaxf(s, 0.0f)
                                  : PI_F / (1.0f + expf(-s));
        xp[(size_t)b * 24 + col_off + i] = o;
    }
}

// ============================================================================
// Kernel 3: 8-qubit state-vector simulator, one wave32 per batch element.
// Amplitude index i in [0,256): lane = i>>3, local j = i&7.
// Wire w -> bit (7-w): bits>=3 cross-lane (shfl_xor), bits<3 in-register.
// ============================================================================
__device__ __forceinline__ float2 cmul(float2 a, float2 b) {
    return make_float2(a.x * b.x - a.y * b.y, a.x * b.y + a.y * b.x);
}
__device__ __forceinline__ float2 cadd(float2 a, float2 b) {
    return make_float2(a.x + b.x, a.y + b.y);
}
__device__ __forceinline__ float2 shfl_xor2(float2 v, int mask) {
    return make_float2(__shfl_xor(v.x, mask, 32), __shfl_xor(v.y, mask, 32));
}

__device__ __forceinline__ void apply_1q(float2* st, int lane, int wire,
                                         float2 u00, float2 u01,
                                         float2 u10, float2 u11)
{
    const int bitpos = 7 - wire;
    if (bitpos >= 3) {
        const int lm    = 1 << (bitpos - 3);
        const int mybit = (lane >> (bitpos - 3)) & 1;
        #pragma unroll
        for (int j = 0; j < 8; ++j) {
            float2 p  = shfl_xor2(st[j], lm);
            float2 r0 = cadd(cmul(u00, st[j]), cmul(u01, p));
            float2 r1 = cadd(cmul(u10, p), cmul(u11, st[j]));
            st[j] = mybit ? r1 : r0;
        }
    } else {
        const int jm = 1 << bitpos;
        #pragma unroll
        for (int j = 0; j < 8; ++j) {
            if ((j & jm) == 0) {
                float2 a0 = st[j], a1 = st[j | jm];
                st[j]      = cadd(cmul(u00, a0), cmul(u01, a1));
                st[j | jm] = cadd(cmul(u10, a0), cmul(u11, a1));
            }
        }
    }
}

// exp(-i t/2 X⊗X): st' = c*st - i*s*flip(wa)flip(wb)(st)
__device__ __forceinline__ void apply_xx(float2* st, int lane, int wa, int wb,
                                         float c, float s)
{
    const int ba = 7 - wa, bb = 7 - wb;
    int lm = 0, jm = 0;
    if (ba >= 3) lm |= 1 << (ba - 3); else jm |= 1 << ba;
    if (bb >= 3) lm |= 1 << (bb - 3); else jm |= 1 << bb;
    float2 nw[8];
    #pragma unroll
    for (int j = 0; j < 8; ++j) {
        float2 f = st[j ^ jm];
        if (lm) f = shfl_xor2(f, lm);
        nw[j] = make_float2(c * st[j].x + s * f.y, c * st[j].y - s * f.x);
    }
    #pragma unroll
    for (int j = 0; j < 8; ++j) st[j] = nw[j];
}

// exp(-i t/2 Z⊗Z): per-amplitude phase (c, -s*zz), zz=+1 iff bits equal
__device__ __forceinline__ void apply_zz(float2* st, int lane, int wa, int wb,
                                         float c, float s)
{
    const int ba = 7 - wa, bb = 7 - wb;
    #pragma unroll
    for (int j = 0; j < 8; ++j) {
        const int i = lane * 8 + j;
        const float zz = (((i >> ba) & 1) == ((i >> bb) & 1)) ? 1.0f : -1.0f;
        st[j] = cmul(st[j], make_float2(c, -s * zz));
    }
}

__global__ __launch_bounds__(256) void quantum_kernel(
    const float* __restrict__ xp,          // [B, 24]
    const float* __restrict__ rot_params,  // [4, 8]
    const float* __restrict__ enta_params, // [4, 8]
    const float* __restrict__ reg_w,       // [8]
    const float* __restrict__ reg_b,       // [1]
    float* __restrict__ out, int B)
{
    const int lane = threadIdx.x & 31;
    const int wave = threadIdx.x >> 5;
    const int b = blockIdx.x * (blockDim.x >> 5) + wave;
    if (b >= B) return;

    // |0...0>
    float2 st[8];
    #pragma unroll
    for (int j = 0; j < 8; ++j) st[j] = make_float2(0.0f, 0.0f);
    st[0].x = (lane == 0) ? 1.0f : 0.0f;

    // data-reuploading angles: qubit q uses xp[b, 3q .. 3q+2]
    float xq[24];
    #pragma unroll
    for (int k = 0; k < 24; ++k) xq[k] = xp[(size_t)b * 24 + k];

    #pragma unroll
    for (int l = 0; l < N_LAYERS; ++l) {
        // ---- 8 batched Rot gates: RZ(omega) RY(theta) RZ(phi) ----
        #pragma unroll
        for (int q = 0; q < N_QUBITS; ++q) {
            const float phi = xq[3 * q], th = xq[3 * q + 1], om = xq[3 * q + 2];
            const float ch = cosf(0.5f * th), sh = sinf(0.5f * th);
            const float aa = 0.5f * (phi + om), bm = 0.5f * (phi - om);
            const float ca = cosf(aa), sa = sinf(aa);
            const float cb = cosf(bm), sb = sinf(bm);
            apply_1q(st, lane, q,
                     make_float2( ca * ch, -sa * ch),   // ep*c
                     make_float2(-cb * sh, -sb * sh),   // -conj(em)*s
                     make_float2( cb * sh, -sb * sh),   // em*s
                     make_float2( ca * ch,  sa * ch));  // conj(ep)*c
        }
        // ---- trainable Rx/Ry + entangler per qubit ----
        #pragma unroll
        for (int q = 0; q < N_QUBITS; ++q) {
            const float t  = rot_params[l * N_QUBITS + q];
            const float ct = cosf(0.5f * t), stv = sinf(0.5f * t);
            if (((l + q) & 1) == 0) {   // Rx
                apply_1q(st, lane, q,
                         make_float2(ct, 0.0f), make_float2(0.0f, -stv),
                         make_float2(0.0f, -stv), make_float2(ct, 0.0f));
            } else {                    // Ry
                apply_1q(st, lane, q,
                         make_float2(ct, 0.0f), make_float2(-stv, 0.0f),
                         make_float2(stv, 0.0f), make_float2(ct, 0.0f));
            }
            const int wa = q, wb = (q + 1) & 7;
            const float p  = enta_params[l * N_QUBITS + q];
            const float cp = cosf(0.5f * p), sp = sinf(0.5f * p);
            if (((l + q) & 1) == 0) apply_xx(st, lane, wa, wb, cp, sp);
            else                    apply_zz(st, lane, wa, wb, cp, sp);
        }
    }

    // PauliZ expvals folded into reg_w dot product, per-lane then wave-reduce
    float rw[8];
    #pragma unroll
    for (int w = 0; w < 8; ++w) rw[w] = reg_w[w];
    float acc = 0.0f;
    #pragma unroll
    for (int j = 0; j < 8; ++j) {
        const int i = lane * 8 + j;
        const float p = st[j].x * st[j].x + st[j].y * st[j].y;
        #pragma unroll
        for (int w = 0; w < 8; ++w) {
            const float sgn = ((i >> (7 - w)) & 1) ? -1.0f : 1.0f;
            acc += p * sgn * rw[w];
        }
    }
    #pragma unroll
    for (int m = 16; m > 0; m >>= 1) acc += __shfl_xor(acc, m, 32);
    if (lane == 0) out[b] = tanhf(acc + reg_b[0]) * 3.0f;
}

// ============================================================================
extern "C" void kernel_launch(void* const* d_in, const int* in_sizes, int n_in,
                              void* d_out, int out_size, void* d_ws, size_t ws_size,
                              hipStream_t stream)
{
    const float* x_a   = (const float*)d_in[0];
    const float* x_v   = (const float*)d_in[1];
    const float* x_t   = (const float*)d_in[2];
    const float* Wih_a = (const float*)d_in[3];
    const float* Whh_a = (const float*)d_in[4];
    const float* bih_a = (const float*)d_in[5];
    const float* bhh_a = (const float*)d_in[6];
    const float* Wih_v = (const float*)d_in[7];
    const float* Whh_v = (const float*)d_in[8];
    const float* bih_v = (const float*)d_in[9];
    const float* bhh_v = (const float*)d_in[10];
    const float* Wih_t = (const float*)d_in[11];
    const float* Whh_t = (const float*)d_in[12];
    const float* bih_t = (const float*)d_in[13];
    const float* bhh_t = (const float*)d_in[14];
    const float* Wp_a  = (const float*)d_in[15];
    const float* bp_a  = (const float*)d_in[16];
    const float* Wp_v  = (const float*)d_in[17];
    const float* bp_v  = (const float*)d_in[18];
    const float* Wp_t  = (const float*)d_in[19];
    const float* bp_t  = (const float*)d_in[20];
    const float* rotp  = (const float*)d_in[21];
    const float* entp  = (const float*)d_in[22];
    const float* reg_w = (const float*)d_in[23];
    const float* reg_b = (const float*)d_in[24];
    float* out = (float*)d_out;

    const int A_IN = 74, V_IN = 35, T_IN = 300;
    const int B = in_sizes[0] / (TT * A_IN);
    const int M = B * TT;

    char* ws = (char*)d_ws;
    const size_t usz = (size_t)M * HID * sizeof(float);
    float* u_a = (float*)(ws);
    float* u_v = (float*)(ws + usz);
    float* u_t = (float*)(ws + 2 * usz);
    float* xp  = (float*)(ws + 3 * usz);

    dim3 blk(256);
    const int gp = (M + 127) / 128;   // 8 waves/block * 16 rows/wave
    proj_wmma_kernel<<<gp, blk, 0, stream>>>(x_a, Wih_a, bih_a, u_a, M, A_IN);
    proj_wmma_kernel<<<gp, blk, 0, stream>>>(x_v, Wih_v, bih_v, u_v, M, V_IN);
    proj_wmma_kernel<<<gp, blk, 0, stream>>>(x_t, Wih_t, bih_t, u_t, M, T_IN);

    const int gr = (B + 255) / 256;
    rnn_proj_kernel<<<gr, blk, 0, stream>>>(u_a, Whh_a, bhh_a, Wp_a, bp_a, xp, 0,  0, B);
    rnn_proj_kernel<<<gr, blk, 0, stream>>>(u_v, Whh_v, bhh_v, Wp_v, bp_v, xp, 1,  8, B);
    rnn_proj_kernel<<<gr, blk, 0, stream>>>(u_t, Whh_t, bhh_t, Wp_t, bp_t, xp, 1, 16, B);

    const int gq = (B + 7) / 8;       // one wave32 per batch element
    quantum_kernel<<<gq, blk, 0, stream>>>(xp, rotp, entp, reg_w, reg_b, out, B);
}
